// SSI_DDI_71004399337987
// MI455X (gfx1250) — matrix operand order repaired and checked by reading, not compiled, not run on hardware
//
#include <hip/hip_runtime.h>
#include <math.h>

// ---------------- problem constants ----------------
constexpr int N_NODES  = 131072;     // 2048 * 64
constexpr int N_EDGES  = 524288;
constexpr int EP       = N_EDGES + N_NODES;  // edges + self loops = 655360
constexpr int B_GRAPHS = 2048;
constexpr int NPG      = 64;
constexpr int F_IN     = 70;
constexpr int D        = 64;         // HEADS*HOUT
constexpr float EPS_SM = 1e-16f;

typedef __attribute__((ext_vector_type(16))) _Float16 v16h;
typedef __attribute__((ext_vector_type(8)))  float    v8f;

// ---------------- helpers ----------------
__device__ __forceinline__ float warpSum(float v) {
  #pragma unroll
  for (int m = 16; m > 0; m >>= 1) v += __shfl_xor(v, m, 32);
  return v;
}

__device__ __forceinline__ void atomicMaxF(float* addr, float v) {
  if (v >= 0.f) atomicMax((int*)addr, __float_as_int(v));
  else          atomicMin((unsigned int*)addr, (unsigned int)__float_as_int(v));
}

// ---------------- generic fill ----------------
__global__ void fill_kernel(float* __restrict__ p, float v, int n) {
  int i = blockIdx.x * 256 + threadIdx.x;
  if (i < n) p[i] = v;
}

// ---------------- initial layernorm over 70 feats (1 wave / node) ----------
__global__ void ln70_kernel(const float* __restrict__ x, const float* __restrict__ w,
                            const float* __restrict__ b, float* __restrict__ y) {
  int lane = threadIdx.x & 31, wv = threadIdx.x >> 5;
  size_t n = (size_t)blockIdx.x * 8 + wv;
  const float* xr = x + n * F_IN;
  float a0 = xr[lane], a1 = xr[lane + 32];
  float a2 = (lane < 6) ? xr[lane + 64] : 0.f;
  float mu = warpSum(a0 + a1 + a2) * (1.0f / 70.0f);
  float d0 = a0 - mu, d1 = a1 - mu, d2 = (lane < 6) ? (a2 - mu) : 0.f;
  float var = warpSum(d0 * d0 + d1 * d1 + d2 * d2) * (1.0f / 70.0f);
  float inv = rsqrtf(var + 1e-5f);
  float* yr = y + n * F_IN;
  yr[lane]      = d0 * inv * w[lane]      + b[lane];
  yr[lane + 32] = d1 * inv * w[lane + 32] + b[lane + 32];
  if (lane < 6) yr[lane + 64] = d2 * inv * w[lane + 64] + b[lane + 64];
}

// ---------------- WMMA GEMM: yl = x@Wl + bl, yr = x@Wr + br ----------------
// x: [N, ldx] f32 (in_dim valid cols), W: [in_dim, 64] f32 row-major.
// Each workgroup: 32 rows x 64 cols; 8 waves each own one 16x16 tile and
// accumulate both outputs (A-fragment reuse across Wl/Wr).
#define LDSK 104
__global__ __launch_bounds__(256) void gemm_xlxr_kernel(
    const float* __restrict__ x, int ldx, int in_dim, int kp,
    const float* __restrict__ Wl, const float* __restrict__ bl,
    const float* __restrict__ Wr, const float* __restrict__ br,
    float* __restrict__ yl, float* __restrict__ yr) {
  __shared__ _Float16 As[32][LDSK];
  __shared__ _Float16 Bls[64][LDSK];   // transposed: [n][k]
  __shared__ _Float16 Brs[64][LDSK];

  int tid = threadIdx.x;
  int rowBase = blockIdx.x * 32;

  // stage W (f32 -> f16, transposed, zero-padded K)
  for (int idx = tid; idx < kp * 64; idx += 256) {
    int k = idx >> 6, n = idx & 63;
    float vl = (k < in_dim) ? Wl[idx] : 0.f;
    float vr = (k < in_dim) ? Wr[idx] : 0.f;
    Bls[n][k] = (_Float16)vl;
    Brs[n][k] = (_Float16)vr;
  }
  // stage A (f32 -> f16, zero-padded K)
  for (int idx = tid; idx < 32 * kp; idx += 256) {
    int r = idx / kp, k = idx % kp;
    float v = (k < in_dim) ? x[(size_t)(rowBase + r) * ldx + k] : 0.f;
    As[r][k] = (_Float16)v;
  }
  __syncthreads();

  int lane = tid & 31;
  int wv   = tid >> 5;
  int subM = wv >> 2, subN = wv & 3;
  int half = lane >> 4, lrow = lane & 15;
  int arow = subM * 16 + lrow;
  int bcol = subN * 16 + lrow;

  v8f accl = {}; v8f accr = {};
  for (int k0 = 0; k0 < kp; k0 += 32) {
    union { v16h v; unsigned int u[8]; } a, b_l, b_r;
    #pragma unroll
    for (int q = 0; q < 4; ++q) {
      // 16-bit A 16x32 layout: e0..7 -> K = half*8 + e ; e8..15 -> K = 16 + half*8 + e
      a.u[q]     = *(const unsigned int*)&As[arow][k0 + half * 8 + 2 * q];
      a.u[q + 4] = *(const unsigned int*)&As[arow][k0 + 16 + half * 8 + 2 * q];
    }
    #pragma unroll
    for (int q = 0; q < 8; ++q) {
      // 16-bit B 32x16 layout: e0..7 -> K = half*8 + e ; e8..15 -> K = 16 + half*8 + e
      int kk2 = k0 + half * 8 + 2 * q + ((q >= 4) ? 8 : 0);
      b_l.u[q] = *(const unsigned int*)&Bls[bcol][kk2];
      b_r.u[q] = *(const unsigned int*)&Brs[bcol][kk2];
    }
    accl = __builtin_amdgcn_wmma_f32_16x16x32_f16(false, a.v, false, b_l.v,
                                                  (short)0, accl, false, false);
    accr = __builtin_amdgcn_wmma_f32_16x16x32_f16(false, a.v, false, b_r.v,
                                                  (short)0, accr, false, false);
  }
  // C/D layout: lane -> col = l%16 ; VGPR v -> row = (l/16)*8 + v
  int col = subN * 16 + lrow;
  #pragma unroll
  for (int v = 0; v < 8; ++v) {
    int orow = rowBase + subM * 16 + half * 8 + v;
    yl[(size_t)orow * D + col] = accl[v] + bl[col];
    yr[(size_t)orow * D + col] = accr[v] + br[col];
  }
}

// ---------------- edge attention logits + per-target max (1 wave / edge) ---
__global__ void logit_kernel(const float* __restrict__ xl, const float* __restrict__ xr,
                             const int* __restrict__ src, const int* __restrict__ dst,
                             const float* __restrict__ att,
                             float* __restrict__ elog, float* __restrict__ nmax) {
  int lane = threadIdx.x & 31, w = threadIdx.x >> 5;
  int e = blockIdx.x * 8 + w;
  int s, d;
  if (e < N_EDGES) { s = src[e]; d = dst[e]; } else { s = d = e - N_EDGES; }
  const float* pl = xl + (size_t)s * D;
  const float* pr = xr + (size_t)d * D;
  float v0 = pl[lane] + pr[lane];
  float v1 = pl[lane + 32] + pr[lane + 32];
  v0 = (v0 > 0.f) ? v0 : 0.2f * v0;
  v1 = (v1 > 0.f) ? v1 : 0.2f * v1;
  v0 *= att[lane];
  v1 *= att[lane + 32];
  v0 = warpSum(v0);
  v1 = warpSum(v1);
  if (lane == 0) {
    elog[(size_t)e * 2]     = v0;
    elog[(size_t)e * 2 + 1] = v1;
    atomicMaxF(&nmax[(size_t)d * 2],     v0);
    atomicMaxF(&nmax[(size_t)d * 2 + 1], v1);
  }
}

// ---------------- exp(logit - max) + denominator -----------------
__global__ void expdenom_kernel(float* __restrict__ elog, const float* __restrict__ nmax,
                                float* __restrict__ nden,
                                const int* __restrict__ dst) {
  int e = blockIdx.x * 256 + threadIdx.x;
  int d = (e < N_EDGES) ? dst[e] : (e - N_EDGES);
  float t0 = expf(elog[(size_t)e * 2]     - nmax[(size_t)d * 2]);
  float t1 = expf(elog[(size_t)e * 2 + 1] - nmax[(size_t)d * 2 + 1]);
  elog[(size_t)e * 2]     = t0;
  elog[(size_t)e * 2 + 1] = t1;
  atomicAdd(&nden[(size_t)d * 2],     t0);
  atomicAdd(&nden[(size_t)d * 2 + 1], t1);
}

// ---------------- weighted message aggregation (1 wave / edge) -------------
__global__ void aggregate_kernel(const float* __restrict__ xl, const float* __restrict__ elog,
                                 const float* __restrict__ nden,
                                 const int* __restrict__ src, const int* __restrict__ dst,
                                 float* __restrict__ xg) {
  int lane = threadIdx.x & 31, w = threadIdx.x >> 5;
  int e = blockIdx.x * 8 + w;
  int s, d;
  if (e < N_EDGES) { s = src[e]; d = dst[e]; } else { s = d = e - N_EDGES; }
  float a0 = elog[(size_t)e * 2]     / (nden[(size_t)d * 2]     + EPS_SM);
  float a1 = elog[(size_t)e * 2 + 1] / (nden[(size_t)d * 2 + 1] + EPS_SM);
  const float* pl = xl + (size_t)s * D;
  float* pd = xg + (size_t)d * D;
  atomicAdd(&pd[lane],      pl[lane]      * a0);
  atomicAdd(&pd[lane + 32], pl[lane + 32] * a1);
}

__global__ void biasadd_kernel(float* __restrict__ xg, const float* __restrict__ bias) {
  int i = blockIdx.x * 256 + threadIdx.x;
  xg[i] += bias[i & 63];
}

// ---------------- GraphConv neighbor sum over original edges ---------------
__global__ void nbr_kernel(const float* __restrict__ xg, const int* __restrict__ src,
                           const int* __restrict__ dst, float* __restrict__ nbr) {
  int lane = threadIdx.x & 31, w = threadIdx.x >> 5;
  int e = blockIdx.x * 8 + w;
  int s = src[e], d = dst[e];
  const float* ps = xg + (size_t)s * D;
  float* pd = nbr + (size_t)d * D;
  atomicAdd(&pd[lane],      ps[lane]);
  atomicAdd(&pd[lane + 32], ps[lane + 32]);
}

// ---------------- SAG score + per-graph max --------------------------------
__global__ void score_kernel(const float* __restrict__ nbr, const float* __restrict__ xg,
                             const float* __restrict__ wrel, const float* __restrict__ brel,
                             const float* __restrict__ wroot,
                             float* __restrict__ score, float* __restrict__ gmax) {
  int lane = threadIdx.x & 31, w = threadIdx.x >> 5;
  int n = blockIdx.x * 8 + w;
  const float* nb = nbr + (size_t)n * D;
  const float* xx = xg + (size_t)n * D;
  float p = nb[lane] * wrel[lane] + nb[lane + 32] * wrel[lane + 32]
          + xx[lane] * wroot[lane] + xx[lane + 32] * wroot[lane + 32];
  p = warpSum(p);
  if (lane == 0) {
    float sc = p + brel[0];
    score[n] = sc;
    atomicMaxF(&gmax[n >> 6], sc);
  }
}

__global__ void snorm_kernel(float* __restrict__ score, const float* __restrict__ gmax,
                             float* __restrict__ gsum) {
  int n = blockIdx.x * 256 + threadIdx.x;
  float s = expf(score[n] - gmax[n >> 6]);
  score[n] = s;
  atomicAdd(&gsum[n >> 6], s);
}

// ---------------- graph embedding: add_pool + mean_pool (cnt == 64) --------
__global__ __launch_bounds__(64) void gemb_kernel(const float* __restrict__ xg,
                                                  const float* __restrict__ sexp,
                                                  const float* __restrict__ gsum,
                                                  float* __restrict__ R, int blk) {
  int g = blockIdx.x, c = threadIdx.x;
  float denom = gsum[g] + EPS_SM;
  float acc = 0.f;
  for (int i = 0; i < NPG; ++i) {
    int n = g * NPG + i;
    acc += xg[(size_t)n * D + c] * (sexp[n] / denom);
  }
  R[((size_t)g * 4 + blk) * D + c] = acc * (65.0f / 64.0f);  // sum + sum/64
}

// ---------------- layernorm(D) + exact GELU --------------------------------
__global__ void ln_gelu_kernel(const float* __restrict__ xg, const float* __restrict__ w,
                               const float* __restrict__ b, float* __restrict__ y) {
  int lane = threadIdx.x & 31, wv = threadIdx.x >> 5;
  size_t n = (size_t)blockIdx.x * 8 + wv;
  const float* x = xg + n * D;
  float a0 = x[lane], a1 = x[lane + 32];
  float mu = warpSum(a0 + a1) * (1.0f / 64.0f);
  float d0 = a0 - mu, d1 = a1 - mu;
  float var = warpSum(d0 * d0 + d1 * d1) * (1.0f / 64.0f);
  float inv = rsqrtf(var + 1e-5f);
  float y0 = d0 * inv * w[lane]      + b[lane];
  float y1 = d1 * inv * w[lane + 32] + b[lane + 32];
  y0 = 0.5f * y0 * (1.f + erff(y0 * 0.70710678118f));
  y1 = 0.5f * y1 * (1.f + erff(y1 * 0.70710678118f));
  float* out = y + n * D;
  out[lane] = y0;
  out[lane + 32] = y1;
}

// ---------------- co-attention + RESCAL readout (1 WG / graph) -------------
__global__ __launch_bounds__(64) void final_kernel(
    const float* __restrict__ Rh, const float* __restrict__ Rt,
    const int* __restrict__ rels,
    const float* __restrict__ Wq, const float* __restrict__ Wk,
    const float* __restrict__ cb, const float* __restrict__ ca,
    const float* __restrict__ rel_emb, float* __restrict__ out) {
  int g = blockIdx.x, tid = threadIdx.x;
  __shared__ float sh[4][64], st[4][64], hn[4][64], tn[4][64], u[4][64];
  __shared__ float red[64];
  __shared__ float sc[16], at[16];
  __shared__ float kk[4][32], qq[4][32];
  __shared__ float rownorm[8];
  __shared__ float rnorm;

  for (int i = 0; i < 4; ++i) {
    sh[i][tid] = Rh[((size_t)g * 4 + i) * D + tid];
    st[i][tid] = Rt[((size_t)g * 4 + i) * D + tid];
  }
  __syncthreads();
  if (tid < 8) {
    const float* row = (tid < 4) ? sh[tid] : st[tid - 4];
    float s = 0.f;
    for (int e = 0; e < 64; ++e) s += row[e] * row[e];
    rownorm[tid] = fmaxf(sqrtf(s), 1e-12f);
  }
  const float* re = rel_emb + (size_t)rels[g] * (D * D);
  float p = 0.f;
  for (int e = 0; e < 64; ++e) { float v = re[tid * 64 + e]; p += v * v; }
  red[tid] = p;
  __syncthreads();
  if (tid == 0) {
    float s = 0.f;
    for (int i = 0; i < 64; ++i) s += red[i];
    rnorm = fmaxf(sqrtf(s), 1e-12f);
  }
  __syncthreads();
  for (int i = 0; i < 4; ++i) {
    hn[i][tid] = sh[i][tid] / rownorm[i];
    tn[i][tid] = st[i][tid] / rownorm[4 + i];
  }
  __syncthreads();
  // u[j][d] = sum_e (rel[d][e]/|rel|) * tn[j][e], d = tid
  for (int j = 0; j < 4; ++j) {
    float s = 0.f;
    for (int e = 0; e < 64; ++e) s += re[tid * 64 + e] * tn[j][e];
    u[j][tid] = s / rnorm;
  }
  __syncthreads();
  // sc[i][j] = hn[i] . u[j]
  for (int pair = 0; pair < 16; ++pair) {
    int i = pair >> 2, j = pair & 3;
    red[tid] = hn[i][tid] * u[j][tid];
    __syncthreads();
    for (int off = 32; off > 0; off >>= 1) {
      if (tid < off) red[tid] += red[tid + off];
      __syncthreads();
    }
    if (tid == 0) sc[pair] = red[0];
    __syncthreads();
  }
  // keys from Rh, queries from Rt
  if (tid < 32) {
    for (int i = 0; i < 4; ++i) {
      float sk = 0.f, sq = 0.f;
      for (int e = 0; e < 64; ++e) {
        sk += sh[i][e] * Wk[e * 32 + tid];
        sq += st[i][e] * Wq[e * 32 + tid];
      }
      kk[i][tid] = sk;
      qq[i][tid] = sq;
    }
  }
  __syncthreads();
  for (int pair = 0; pair < 16; ++pair) {
    int i = pair >> 2, j = pair & 3;
    red[tid] = (tid < 32) ? tanhf(qq[j][tid] + kk[i][tid] + cb[tid]) * ca[tid] : 0.f;
    __syncthreads();
    for (int off = 32; off > 0; off >>= 1) {
      if (tid < off) red[tid] += red[tid + off];
      __syncthreads();
    }
    if (tid == 0) at[pair] = red[0];
    __syncthreads();
  }
  if (tid == 0) {
    float s = 0.f;
    for (int q = 0; q < 16; ++q) s += at[q] * sc[q];
    out[g] = s;
  }
}

// ---------------- host driver ----------------
extern "C" void kernel_launch(void* const* d_in, const int* in_sizes, int n_in,
                              void* d_out, int out_size, void* d_ws, size_t ws_size,
                              hipStream_t stream) {
  (void)in_sizes; (void)n_in; (void)out_size; (void)ws_size;
  // setup_inputs() dict order, params flattened in insertion order:
  const float* xh   = (const float*)d_in[0];
  const float* xt   = (const float*)d_in[1];
  const int*   eh   = (const int*)d_in[2];   // [2, E]
  const int*   et   = (const int*)d_in[3];
  const int*   rels = (const int*)d_in[5];   // d_in[4] = batch (implicit: n/64)
  const float* init_w = (const float*)d_in[6];
  const float* init_b = (const float*)d_in[7];
  const float* co_Wq = (const float*)d_in[52];
  const float* co_Wk = (const float*)d_in[53];
  const float* co_b  = (const float*)d_in[54];
  const float* co_a  = (const float*)d_in[55];
  const float* rel_emb = (const float*)d_in[56];

  char* ws = (char*)d_ws;
  size_t off = 0;
  auto alloc = [&](size_t bytes) {
    void* p = ws + off;
    off = (off + bytes + 255) & ~(size_t)255;
    return p;
  };
  float* bufX  = (float*)alloc((size_t)N_NODES * F_IN * 4);
  float* bufXn = (float*)alloc((size_t)N_NODES * D * 4);
  float* xg    = (float*)alloc((size_t)N_NODES * D * 4);
  float* xl    = (float*)alloc((size_t)N_NODES * D * 4);
  float* xr    = (float*)alloc((size_t)N_NODES * D * 4);
  float* nbr   = (float*)alloc((size_t)N_NODES * D * 4);
  float* elog  = (float*)alloc((size_t)EP * 2 * 4);
  float* nmax  = (float*)alloc((size_t)N_NODES * 2 * 4);
  float* nden  = (float*)alloc((size_t)N_NODES * 2 * 4);
  float* sexp  = (float*)alloc((size_t)N_NODES * 4);
  float* gmax  = (float*)alloc((size_t)B_GRAPHS * 4);
  float* gsum  = (float*)alloc((size_t)B_GRAPHS * 4);
  float* Rh    = (float*)alloc((size_t)B_GRAPHS * 4 * D * 4);
  float* Rt    = (float*)alloc((size_t)B_GRAPHS * 4 * D * 4);

  for (int side = 0; side < 2; ++side) {
    const float* xin = side ? xt : xh;
    const int* src = side ? et : eh;
    const int* dst = src + N_EDGES;
    float* R = side ? Rt : Rh;

    ln70_kernel<<<N_NODES / 8, 256, 0, stream>>>(xin, init_w, init_b, bufX);
    const float* cur = bufX;
    int ld = F_IN, indim = F_IN, kp = 96;

    for (int blk = 0; blk < 4; ++blk) {
      int pb = 8 + blk * 11;
      const float* Wl    = (const float*)d_in[pb + 0];
      const float* bl    = (const float*)d_in[pb + 1];
      const float* Wr    = (const float*)d_in[pb + 2];
      const float* br    = (const float*)d_in[pb + 3];
      const float* att   = (const float*)d_in[pb + 4];
      const float* bias  = (const float*)d_in[pb + 5];
      const float* wrel  = (const float*)d_in[pb + 6];
      const float* brel  = (const float*)d_in[pb + 7];
      const float* wroot = (const float*)d_in[pb + 8];
      const float* nw    = (const float*)d_in[pb + 9];
      const float* nb    = (const float*)d_in[pb + 10];

      fill_kernel<<<(N_NODES * D) / 256, 256, 0, stream>>>(xg, 0.f, N_NODES * D);
      fill_kernel<<<(N_NODES * D) / 256, 256, 0, stream>>>(nbr, 0.f, N_NODES * D);
      fill_kernel<<<(N_NODES * 2) / 256, 256, 0, stream>>>(nden, 0.f, N_NODES * 2);
      fill_kernel<<<(N_NODES * 2) / 256, 256, 0, stream>>>(nmax, -INFINITY, N_NODES * 2);
      fill_kernel<<<(B_GRAPHS + 255) / 256, 256, 0, stream>>>(gsum, 0.f, B_GRAPHS);
      fill_kernel<<<(B_GRAPHS + 255) / 256, 256, 0, stream>>>(gmax, -INFINITY, B_GRAPHS);

      gemm_xlxr_kernel<<<N_NODES / 32, 256, 0, stream>>>(cur, ld, indim, kp,
                                                         Wl, bl, Wr, br, xl, xr);
      logit_kernel<<<EP / 8, 256, 0, stream>>>(xl, xr, src, dst, att, elog, nmax);
      expdenom_kernel<<<EP / 256, 256, 0, stream>>>(elog, nmax, nden, dst);
      aggregate_kernel<<<EP / 8, 256, 0, stream>>>(xl, elog, nden, src, dst, xg);
      biasadd_kernel<<<(N_NODES * D) / 256, 256, 0, stream>>>(xg, bias);
      nbr_kernel<<<N_EDGES / 8, 256, 0, stream>>>(xg, src, dst, nbr);
      score_kernel<<<N_NODES / 8, 256, 0, stream>>>(nbr, xg, wrel, brel, wroot,
                                                    sexp, gmax);
      snorm_kernel<<<N_NODES / 256, 256, 0, stream>>>(sexp, gmax, gsum);
      gemb_kernel<<<B_GRAPHS, 64, 0, stream>>>(xg, sexp, gsum, R, blk);
      ln_gelu_kernel<<<N_NODES / 8, 256, 0, stream>>>(xg, nw, nb, bufXn);
      cur = bufXn; ld = D; indim = D; kp = 64;
    }
  }
  final_kernel<<<B_GRAPHS, 64, 0, stream>>>(Rh, Rt, rels, co_Wq, co_Wk, co_b, co_a,
                                            rel_emb, (float*)d_out);
}
